// SparseMOE_4346506904194
// MI455X (gfx1250) — matrix operand order repaired and checked by reading, not compiled, not run on hardware
//
#include <hip/hip_runtime.h>

#define TOK 16384   // B*S
#define DIM 1024    // D
#define NE  8       // experts
#define TOPK 2
#define BSTRIDE 40  // padded LDS row stride (bf16 elems): spreads 16-lane b128 frag reads over all 64 DS banks

typedef __bf16 bf16;
typedef __attribute__((ext_vector_type(16))) __bf16 v16bf;
typedef __attribute__((ext_vector_type(8)))  __bf16 bf16x8;
typedef __attribute__((ext_vector_type(4)))  __bf16 bf16x4;
typedef __attribute__((ext_vector_type(8)))  float  v8f;
typedef int v4i __attribute__((vector_size(16)));   // matches async-LDS builtin param type

#define AS1 __attribute__((address_space(1)))
#define AS3 __attribute__((address_space(3)))

#if __has_builtin(__builtin_amdgcn_global_load_async_to_lds_b128) && \
    __has_builtin(__builtin_amdgcn_s_wait_asynccnt)
#define USE_ASYNC_LDS 1
#else
#define USE_ASYNC_LDS 0
#endif

#if USE_ASYNC_LDS
// global (AS1) -> LDS (AS3) async 16B copy; flat->LDS offset is a low-32-bit
// truncation per the ISA aperture rules, so integer casts are exact.
#define ASYNC_CP16(gsrc, ldst)                                              \
    __builtin_amdgcn_global_load_async_to_lds_b128(                         \
        (AS1 v4i*)(uintptr_t)(gsrc), (AS3 v4i*)(uint32_t)(uintptr_t)(ldst), \
        0, 0)
#endif

__device__ __forceinline__ v16bf pack16(bf16x8 a, bf16x8 b) {
    return __builtin_shufflevector(a, b, 0, 1, 2, 3, 4, 5, 6, 7,
                                         8, 9, 10, 11, 12, 13, 14, 15);
}

// ---------------- counters ----------------
__global__ void moe_zero_counts(int* counts) {
    if (threadIdx.x < NE) counts[threadIdx.x] = 0;
}

// ---------------- f32 -> bf16 conversion ----------------
__global__ void moe_cvt_bf16(const float* __restrict__ src,
                             bf16* __restrict__ dst, long n) {
    long stride = (long)gridDim.x * blockDim.x * 4;
    for (long i = ((long)blockIdx.x * blockDim.x + threadIdx.x) * 4; i < n; i += stride) {
        float4 v = *(const float4*)(src + i);
        bf16x4 o;
        o[0] = (bf16)v.x; o[1] = (bf16)v.y; o[2] = (bf16)v.z; o[3] = (bf16)v.w;
        *(bf16x4*)(dst + i) = o;
    }
}

// ---------------- router: logits, top-2, per-expert lists ----------------
__global__ void moe_router(const float* __restrict__ x,
                           const float* __restrict__ gw,
                           const float* __restrict__ gb,
                           float* __restrict__ logits_out,
                           int*   __restrict__ sel_out,
                           int*   __restrict__ topk_e,
                           float* __restrict__ topk_w,
                           int*   __restrict__ counts,
                           int*   __restrict__ tlist,
                           float* __restrict__ wlist) {
    int t    = (int)((blockIdx.x * blockDim.x + threadIdx.x) >> 5);  // one wave32 per token
    int lane = threadIdx.x & 31;
    if (t >= TOK) return;

    const float* xr = x + (size_t)t * DIM;
    float acc[NE];
#pragma unroll
    for (int e = 0; e < NE; ++e) acc[e] = 0.f;

    for (int d = lane; d < DIM; d += 32) {
        float xv = xr[d];
#pragma unroll
        for (int e = 0; e < NE; ++e) acc[e] += xv * gw[e * DIM + d];
    }
#pragma unroll
    for (int off = 16; off >= 1; off >>= 1) {
#pragma unroll
        for (int e = 0; e < NE; ++e) acc[e] += __shfl_xor(acc[e], off, 32);
    }

    if (lane == 0) {
        float lg[NE];
#pragma unroll
        for (int e = 0; e < NE; ++e) {
            lg[e] = acc[e] + gb[e];
            logits_out[t * NE + e] = lg[e];
        }
        // top-2 on logits (softmax monotonic; ties -> lowest index, like jax top_k)
        int e1 = 0;
#pragma unroll
        for (int e = 1; e < NE; ++e) if (lg[e] > lg[e1]) e1 = e;
        int e2 = (e1 == 0) ? 1 : 0;
#pragma unroll
        for (int e = 0; e < NE; ++e) if (e != e1 && lg[e] > lg[e2]) e2 = e;

        // renormalized top-2 softmax weights: w1 = p1/(p1+p2)
        float d21 = expf(lg[e2] - lg[e1]);
        float w1  = 1.f / (1.f + d21);
        float w2  = d21 / (1.f + d21);

        sel_out[t * TOPK + 0] = e1;
        sel_out[t * TOPK + 1] = e2;
        topk_e[t * TOPK + 0] = e1;  topk_w[t * TOPK + 0] = w1;
        topk_e[t * TOPK + 1] = e2;  topk_w[t * TOPK + 1] = w2;

        int p1 = atomicAdd(&counts[e1], 1);
        tlist[e1 * TOK + p1] = t;  wlist[e1 * TOK + p1] = w1;
        int p2 = atomicAdd(&counts[e2], 1);
        tlist[e2 * TOK + p2] = t;  wlist[e2 * TOK + p2] = w2;
    }
}

// ---------------- init out with weighted expert biases ----------------
__global__ void moe_init_out(const int* __restrict__ topk_e,
                             const float* __restrict__ topk_w,
                             const float* __restrict__ eb,
                             float* __restrict__ out) {
    long n = (long)TOK * DIM;
    long stride = (long)gridDim.x * blockDim.x;
    for (long i = (long)blockIdx.x * blockDim.x + threadIdx.x; i < n; i += stride) {
        int t = (int)(i >> 10);
        int o = (int)(i & (DIM - 1));
        int e1 = topk_e[t * TOPK], e2 = topk_e[t * TOPK + 1];
        float w1 = topk_w[t * TOPK], w2 = topk_w[t * TOPK + 1];
        out[i] = w1 * eb[e1 * DIM + o] + w2 * eb[e2 * DIM + o];
    }
}

// ---------------- expert GEMM: Y = X_gathered @ W_e^T, weighted scatter ----------------
// Block = 256 thr = 8 waves; tile 128(M) x 64(N); B tile (32K x 64N) double-buffered
// in LDS, staged with async global->LDS (ASYNCcnt path).
__global__ __launch_bounds__(256)
void moe_expert_gemm(const bf16* __restrict__ xh,
                     const bf16* __restrict__ wh,
                     const int*  __restrict__ counts,
                     const int*  __restrict__ tlist,
                     const float* __restrict__ wlist,
                     float* __restrict__ out) {
    __shared__ bf16 Bs[2][64 * BSTRIDE];   // 2 x 5 KB

    const int e   = blockIdx.z;
    const int cnt = counts[e];
    if ((int)blockIdx.y * 128 >= cnt) return;      // block-uniform exit (barrier-safe)

    const int tid  = threadIdx.x;
    const int wave = tid >> 5;
    const int lane = tid & 31;
    const int half = lane >> 4;
    const int l15  = lane & 15;

    const int m_base = blockIdx.y * 128 + wave * 16;
    const int n_base = blockIdx.x * 64;

    const int*   tl = tlist + e * TOK;
    const float* wl = wlist + e * TOK;

    // A row for this lane; clamp (cnt >= 1 guaranteed). Rows >= cnt compute
    // garbage that the guarded epilogue discards — keeps EXEC all-1s at WMMA.
    int rowm = m_base + l15;
    if (rowm >= cnt) rowm = cnt - 1;
    const bf16* aptr  = xh + (size_t)tl[rowm] * DIM;
    const bf16* bbase = wh + (size_t)e * DIM * DIM;

    // Cooperative B staging: 64 rows x 32 k = 4 KB -> 16 B per thread.
    const int bn = tid >> 2;                 // B row 0..63
    const int bc = tid & 3;                  // 16B chunk 0..3 within the 64B k-slice
    const bf16* bg = bbase + (size_t)(n_base + bn) * DIM + bc * 8;
    bf16* bl0 = &Bs[0][bn * BSTRIDE + bc * 8];

#if USE_ASYNC_LDS
    ASYNC_CP16(bg, bl0);
    __builtin_amdgcn_s_wait_asynccnt(0);
    __syncthreads();
#else
    bf16x8 stage = *(const bf16x8*)bg;
    *(bf16x8*)bl0 = stage;
    __syncthreads();
#endif

    v8f acc[4] = {};
    int buf = 0;
    for (int k = 0; k < DIM; k += 32) {
        bf16* blnext = bl0 + (buf ^ 1) * (64 * BSTRIDE);
        if (k + 32 < DIM) {
#if USE_ASYNC_LDS
            ASYNC_CP16(bg + k + 32, blnext);
#else
            stage = *(const bf16x8*)(bg + k + 32);
#endif
        }

        // A 16x32 bf16 fragment: per lane, K chunks [half*8..+7] and [+16..+23]
        int ka = k + half * 8;
        bf16x8 a0 = *(const bf16x8*)(aptr + ka);
        bf16x8 a1 = *(const bf16x8*)(aptr + ka + 16);
        v16bf a = pack16(a0, a1);

        // B 32x16 fragments from LDS: column l15+16j, K chunks [half*8..+7] and [+16..+23]
        const bf16* bsbuf = &Bs[buf][0];
#pragma unroll
        for (int j = 0; j < 4; ++j) {
            const bf16* bp = bsbuf + (l15 + 16 * j) * BSTRIDE + half * 8;
            bf16x8 b0 = *(const bf16x8*)bp;
            bf16x8 b1 = *(const bf16x8*)(bp + 16);
            v16bf b = pack16(b0, b1);
            acc[j] = __builtin_amdgcn_wmma_f32_16x16x32_bf16(
                         false, a, false, b, (short)0, acc[j], false, false);
        }

        if (k + 32 < DIM) {
#if USE_ASYNC_LDS
            __builtin_amdgcn_s_wait_asynccnt(0);   // this wave's async writes landed
#else
            *(bf16x8*)blnext = stage;
#endif
            __syncthreads();                       // publish buf^1 to all waves
            buf ^= 1;
        }
    }

    // C/D layout: lanes 0-15: N=lane, VGPR v -> M=v ; lanes 16-31: N=lane-16, M=v+8
#pragma unroll
    for (int j = 0; j < 4; ++j) {
#pragma unroll
        for (int v = 0; v < 8; ++v) {
            int m = m_base + half * 8 + v;
            if (m < cnt) {
                int   tok = tl[m];
                float w   = wl[m];
                atomicAdd(out + (size_t)tok * DIM + (n_base + 16 * j + l15),
                          w * acc[j][v]);
            }
        }
    }
}

// ---------------- host launcher ----------------
extern "C" void kernel_launch(void* const* d_in, const int* in_sizes, int n_in,
                              void* d_out, int out_size, void* d_ws, size_t ws_size,
                              hipStream_t stream) {
    const float* x  = (const float*)d_in[0];   // [B,S,D]
    const float* gw = (const float*)d_in[1];   // [E,D]
    const float* gb = (const float*)d_in[2];   // [E]
    const float* ew = (const float*)d_in[3];   // [E,D,D]
    const float* eb = (const float*)d_in[4];   // [E,D]

    float* out    = (float*)d_out;                       // [T,D]
    float* logits = out + (size_t)TOK * DIM;             // [T,E]
    int*   sel    = (int*)(logits + (size_t)TOK * NE);   // [T,K] int32

    char* p = (char*)d_ws;
    bf16* xh = (bf16*)p;      p += (size_t)TOK * DIM * 2;        // 32 MB
    bf16* wh = (bf16*)p;      p += (size_t)NE * DIM * DIM * 2;   // 16 MB
    int*  counts = (int*)p;   p += 256;
    int*  tlist  = (int*)p;   p += (size_t)NE * TOK * 4;         // 512 KB
    float* wlist = (float*)p; p += (size_t)NE * TOK * 4;         // 512 KB
    int*  topk_e = (int*)p;   p += (size_t)TOK * TOPK * 4;
    float* topk_w = (float*)p; p += (size_t)TOK * TOPK * 4;

    moe_zero_counts<<<1, 32, 0, stream>>>(counts);
    moe_cvt_bf16<<<2048, 256, 0, stream>>>(x,  xh, (long)TOK * DIM);
    moe_cvt_bf16<<<2048, 256, 0, stream>>>(ew, wh, (long)NE * DIM * DIM);
    moe_router<<<(TOK * 32) / 256, 256, 0, stream>>>(
        x, gw, gb, logits, sel, topk_e, topk_w, counts, tlist, wlist);
    moe_init_out<<<4096, 256, 0, stream>>>(topk_e, topk_w, eb, out);

    dim3 g(DIM / 64, (TOK + 127) / 128, NE);   // worst case: one expert owns all tokens
    moe_expert_gemm<<<g, 256, 0, stream>>>(xh, wh, counts, tlist, wlist, out);
}
